// FeatSimLossV2_41472204210805
// MI455X (gfx1250) — matrix operand order repaired and checked by reading, not compiled, not run on hardware
//
#include <hip/hip_runtime.h>

// ---------------------------------------------------------------------------
// FeatSimLoss fused pipeline for MI455X (gfx1250, wave32).
//  K1 softmax  : probT[pixel][32] fp16 (ch>=19 zeroed) + per-pixel f32 sum
//  K2 loss     : pos via v_wmma_f32_16x16x32_f16 diagonals; fragments loaded
//                as contiguous b128 chunks; branch-free OOB via a zero page
//                adjacent to probT (32-bit offset cndmask); FMA-based diag
//                extraction; per-pixel top-k; wave-reduced atomics
//  K3 finalize : 4 scalar losses
//  K4 states   : mean over 9 channels of sim_feat0
// ---------------------------------------------------------------------------

typedef _Float16 v16h __attribute__((ext_vector_type(16)));
typedef _Float16 v8h  __attribute__((ext_vector_type(8)));
typedef float    v8f  __attribute__((ext_vector_type(8)));

#define B_    4
#define C_    19
#define H_    256
#define W_    256
#define HW_   (H_ * W_)          // 65536
#define NPIX  (B_ * HW_)         // 262144
#define CPAD  32                 // padded channel count (f16) per pixel

union HV16 { v16h v; v8h h[2]; };

// ---------------- K1: softmax over channels --------------------------------
__global__ __launch_bounds__(256) void fsl_softmax_kernel(
    const float* __restrict__ logits,
    _Float16*   __restrict__ probT,     // [NPIX][CPAD] + 32 f16 zero page
    float*      __restrict__ sumP,      // [NPIX]
    float*      __restrict__ acc) {     // 8 accumulators
  const int p = blockIdx.x * 256 + threadIdx.x;
  if (blockIdx.x == 0) {
    if (threadIdx.x < 8) acc[threadIdx.x] = 0.0f;
    else if (threadIdx.x < 24)            // zero page right after probT
      ((float*)(probT + (size_t)NPIX * CPAD))[threadIdx.x - 8] = 0.0f;
  }
  if (p >= NPIX) return;
  const int b  = p >> 16;
  const int hw = p & (HW_ - 1);
  const float* src = logits + (size_t)b * C_ * HW_ + hw;
  float v[C_];
  float m = -3.4e38f;
#pragma unroll
  for (int c = 0; c < C_; ++c) { v[c] = src[(size_t)c * HW_]; m = fmaxf(m, v[c]); }
  float s = 0.0f;
#pragma unroll
  for (int c = 0; c < C_; ++c) { v[c] = __expf(v[c] - m); s += v[c]; }
  const float inv = 1.0f / s;
  float s2 = 0.0f;
  HV16 q[2];
#pragma unroll
  for (int c = 0; c < CPAD; ++c) {
    float qq = 0.0f;
    if (c < C_) { qq = v[c] * inv; s2 += qq; }
    q[c >> 4].v[c & 15] = (_Float16)qq;
  }
  v8h* dst = (v8h*)(probT + (size_t)p * CPAD);
  dst[0] = q[0].h[0];
  dst[1] = q[0].h[1];
  dst[2] = q[1].h[0];
  dst[3] = q[1].h[1];
  sumP[p] = s2;
}

// ---------------- per-scale top-k loss (set membership only) ---------------
template <int SHIFT, int SW>
__device__ __forceinline__ void fsl_scale_loss(
    const float* __restrict__ sim, int b, int h, int w,
    const float* pos, const float* neg,
    float& lp, float& ln, float& m) {
  const int sh = h >> SHIFT, sw = w >> SHIFT;
  const float* sp = sim + (size_t)b * 9 * SW * SW + sh * SW + sw;
  float simt[9];
#pragma unroll
  for (int t = 0; t < 9; ++t) simt[t] = sp[(size_t)t * SW * SW];
  m = (simt[0] > 0.0f) ? 1.0f : 0.0f;
  unsigned top = 0;                       // top-(TOP_K+1) = top-5 membership
#pragma unroll
  for (int j = 0; j < 5; ++j) {
    float best = -3.4e38f; int bi = 0;
#pragma unroll
    for (int t = 0; t < 9; ++t) {
      bool take = (((top >> t) & 1u) == 0u) && (simt[t] > best);  // strict > => lowest idx wins ties
      best = take ? simt[t] : best;
      bi   = take ? t : bi;
    }
    top |= 1u << bi;
  }
  lp = 0.0f; ln = 0.0f;                   // bottom-4 = complement (distinct values)
#pragma unroll
  for (int t = 0; t < 9; ++t) {
    if ((top >> t) & 1u) lp += simt[t] * (-pos[t]);
    else                 ln += (1.0f - simt[t]) * (-neg[t]);
  }
}

// ---------------- K2: WMMA pos + losses ------------------------------------
// One wave = 32 consecutive pixels (two 16-pixel WMMA groups).
__global__ __launch_bounds__(256) void fsl_loss_kernel(
    const _Float16* __restrict__ probT,   // [NPIX][CPAD] + zero page
    const float*    __restrict__ sumP,
    const float*    __restrict__ sim0,
    const float*    __restrict__ sim1,
    float*          __restrict__ acc) {
  const int lane    = threadIdx.x & 31;
  const int wave    = threadIdx.x >> 5;
  const int pixBase = (blockIdx.x * 8 + wave) * 32;
  const int n16     = lane & 15;
  const int hiHalf  = lane >> 4;

  // One-hot diag-extraction weights (loop-invariant, FMA-based select):
  // diag(m,m) lives at lane m / VGPR m (m<8) or lane m+16 / VGPR m-8 (m>=8).
  const int ridx = (lane < 8) ? lane : (lane >= 24 ? lane - 24 : -1);
  float wsel[8];
#pragma unroll
  for (int r = 0; r < 8; ++r) wsel[r] = (ridx == r) ? 1.0f : 0.0f;

  // ---- A fragments: center probs; two contiguous 8-ch chunks per lane -----
  // A 16x32 (contiguous K, matches B): lane<16 holds K 0..15; lane>=16 K 16..31
  v16h afrag[2];
#pragma unroll
  for (int g = 0; g < 2; ++g) {
    const int pix = pixBase + g * 16 + n16;
    const _Float16* bp = probT + (size_t)pix * CPAD;
    HV16 u;
    u.h[0] = *(const v8h*)(bp + 16 * hiHalf);
    u.h[1] = *(const v8h*)(bp + 16 * hiHalf + 8);
    afrag[g] = u.v;
  }

  // ---- per-tap: B fragment (neighbor probs), WMMA, diagonal extract -------
  // B 32x16 (contiguous K, matches A): lane<16 holds K 0..15 of col N=lane; lane>=16 K 16..31
  float pos[9];
#pragma unroll
  for (int t = 0; t < 9; ++t) {
    const int dy = (t / 3) * 2 - 2;
    const int dx = (t % 3) * 2 - 2;
    float diagv[2];
#pragma unroll
    for (int g = 0; g < 2; ++g) {
      const int pg  = pixBase + g * 16;
      const int pb  = pg >> 16;
      const int ph  = (pg >> 8) & 255;
      const int pw0 = pg & 255;
      const int nh  = ph + dy;
      const int nw  = pw0 + n16 + dx;
      const bool valid = (nh >= 0) & (nh < H_) & (nw >= 0) & (nw < W_);
      const int nPix = (pb << 16) + (nh << 8) + nw;
      // branch-free OOB: 32-bit element offset select into probT / zero page
      const int off = valid ? nPix * CPAD : NPIX * CPAD;
      const _Float16* nbp = probT + off + 16 * hiHalf;
      HV16 ub;
      ub.h[0] = *(const v8h*)(nbp);
      ub.h[1] = *(const v8h*)(nbp + 8);
      v8f cz = {0.f, 0.f, 0.f, 0.f, 0.f, 0.f, 0.f, 0.f};
      v8f d  = __builtin_amdgcn_wmma_f32_16x16x32_f16(false, afrag[g], false, ub.v,
                                                      (short)0, cz, false, false);
      float md = 0.0f;
#pragma unroll
      for (int r = 0; r < 8; ++r) md = fmaf(d[r], wsel[r], md);
      diagv[g] = md;
    }
    const int src = (n16 < 8) ? n16 : (n16 + 16);
    const float p0 = __shfl(diagv[0], src, 32);
    const float p1 = __shfl(diagv[1], src, 32);
    pos[t] = (hiHalf == 0) ? p0 : p1;                      // lane l <-> pixel pixBase+l
  }

  // ---- scalar per-pixel phase --------------------------------------------
  const int myPix = pixBase + lane;
  const int b  = myPix >> 16;
  const int h  = (myPix >> 8) & 255;
  const int w  = myPix & 255;
  const float sc = sumP[myPix];

  float neg[9];
#pragma unroll
  for (int t = 0; t < 9; ++t) {
    const int dy = (t / 3) * 2 - 2;
    const int dx = (t % 3) * 2 - 2;
    const int nh = h + dy, nw = w + dx;
    const bool valid = (nh >= 0) & (nh < H_) & (nw >= 0) & (nw < W_);
    const int nhc = min(max(nh, 0), H_ - 1);
    const int nwc = min(max(nw, 0), W_ - 1);
    float sq = sumP[(b << 16) + (nhc << 8) + nwc];         // clamped, branch-free
    sq = valid ? sq : 0.0f;
    neg[t] = sc * sq - pos[t];
  }

  float lp0, ln0, m0, lp1, ln1, m1;
  fsl_scale_loss<2, 64 >(sim0, b, h, w, pos, neg, lp0, ln0, m0);
  fsl_scale_loss<1, 128>(sim1, b, h, w, pos, neg, lp1, ln1, m1);

  float vals[6] = {lp0 * m0, ln0 * m0, m0, lp1 * m1, ln1 * m1, m1};
#pragma unroll
  for (int i = 0; i < 6; ++i) {
    float v = vals[i];
#pragma unroll
    for (int off = 16; off > 0; off >>= 1) v += __shfl_down(v, off, 32);
    if (lane == 0) atomicAdd(&acc[i], v);
  }
}

// ---------------- K3: finalize losses --------------------------------------
__global__ void fsl_finalize_kernel(const float* __restrict__ acc,
                                    float* __restrict__ out) {
  if (threadIdx.x == 0 && blockIdx.x == 0) {
    const float c0 = acc[2], c1 = acc[5];
    out[0] = acc[0] / (5.0f * c0) * 1.0f;   // loss_pos_0 (w=1)
    out[1] = acc[1] / (4.0f * c0) * 1.0f;   // loss_neg_0
    out[2] = acc[3] / (5.0f * c1) * 1.0f;   // loss_pos_1
    out[3] = acc[4] / (4.0f * c1) * 1.0f;   // loss_neg_1
  }
}

// ---------------- K4: states_sim = mean(sim_feat0, ch) ---------------------
__global__ __launch_bounds__(256) void fsl_states_kernel(
    const float* __restrict__ sim0, float* __restrict__ out) {
  int i = blockIdx.x * 256 + threadIdx.x;       // 4*64*64 = 16384
  if (i >= B_ * 4096) return;
  const int b  = i >> 12;
  const int hw = i & 4095;
  const float* sp = sim0 + (size_t)b * 9 * 4096 + hw;
  float s = 0.0f;
#pragma unroll
  for (int c = 0; c < 9; ++c) s += sp[(size_t)c * 4096];
  out[4 + i] = s * (1.0f / 9.0f);
}

// ---------------------------------------------------------------------------
extern "C" void kernel_launch(void* const* d_in, const int* in_sizes, int n_in,
                              void* d_out, int out_size, void* d_ws, size_t ws_size,
                              hipStream_t stream) {
  (void)in_sizes; (void)n_in; (void)out_size; (void)ws_size;
  const float* seg  = (const float*)d_in[0];   // [4,19,256,256]
  const float* sim0 = (const float*)d_in[1];   // [4,9,64,64]
  const float* sim1 = (const float*)d_in[2];   // [4,9,128,128]
  float* out = (float*)d_out;                  // [4 losses | 4*64*64 states]

  char* ws = (char*)d_ws;                      // ~18 MB used
  float*    acc   = (float*)ws;                                        // 8 f32 @ 0
  _Float16* probT = (_Float16*)(ws + 256);                             // NPIX*CPAD f16
  // 64B zero page lives at probT + NPIX*CPAD (reached via offset select)
  float*    sumP  = (float*)(ws + 256 + sizeof(_Float16) * ((size_t)CPAD * NPIX + 64));

  fsl_softmax_kernel<<<NPIX / 256, 256, 0, stream>>>(seg, probT, sumP, acc);
  fsl_loss_kernel   <<<NPIX / 256, 256, 0, stream>>>(probT, sumP, sim0, sim1, acc);
  fsl_finalize_kernel<<<1, 32, 0, stream>>>(acc, out);
  fsl_states_kernel <<<(B_ * 4096) / 256, 256, 0, stream>>>(sim0, out);
}
